// MultiheadAttentionCustom_12936441495921
// MI455X (gfx1250) — matrix operand order repaired and checked by reading, not compiled, not run on hardware
//
#include <hip/hip_runtime.h>

// Problem constants (from reference): B=2, S=2048, D=1024, H=16, DH=64
constexpr int Bn   = 2;
constexpr int Sdim = 2048;
constexpr int Ddim = 1024;
constexpr int Hn   = 16;
constexpr int DHn  = 64;
constexpr int Msz  = Bn * Sdim;       // 4096 rows for the projection GEMMs
constexpr float NEGV = -10000.0f;

typedef __attribute__((ext_vector_type(16))) _Float16 v16h;
typedef __attribute__((ext_vector_type(8)))  float    v8f;
typedef __attribute__((ext_vector_type(4)))  unsigned v4u;
typedef __attribute__((ext_vector_type(8)))  int      v8i;
typedef __attribute__((ext_vector_type(4)))  int      v4i;

// Feature probes (safe fallbacks keep the kernel compiling on any toolchain).
#if defined(__has_builtin)
#  if __has_builtin(__builtin_amdgcn_tensor_load_to_lds)
#    define HAS_TDM 1
#  endif
#  if __has_builtin(__builtin_amdgcn_global_load_async_to_lds_b128)
#    define HAS_ASYNC 1
#  endif
#  if __has_builtin(__builtin_amdgcn_s_wait_asynccnt)
#    define HAS_WAIT_ASYNC 1
#  endif
#endif
#ifndef HAS_TDM
#define HAS_TDM 0
#endif
#ifndef HAS_ASYNC
#define HAS_ASYNC 0
#endif
#ifndef HAS_WAIT_ASYNC
#define HAS_WAIT_ASYNC 0
#endif

union U16 {                 // one WMMA 16-bit operand fragment (8 VGPRs)
    v16h v;
    unsigned u[8];
    uint4 q[2];
    _Float16 h[16];
};
union H4 {                  // 4 halves <-> 8 bytes for vectorized LDS stores
    _Float16 h[4];
    uint2 u;
};

#if HAS_TDM
// ---------------------------------------------------------------------------
// Tensor Data Mover: DMA a 64x64 f16 tile (row stride `rowStride` elements)
// from global into LDS at byte offset `ldsoff`, padding +4 dwords every
// 32 dwords so LDS rows land on a 72-half (144 B) stride.
// D# bit layout per CDNA5 ISA 8.3/8.4. Issued by one wave; EXEC is ignored.
// ---------------------------------------------------------------------------
__device__ __forceinline__ void tdm_load_tile64(
    const _Float16* gsrc, unsigned ldsoff, unsigned rowStride,
    unsigned dim0, unsigned dim1)
{
    unsigned long long ga = (unsigned long long)(size_t)(const void*)gsrc;
    v4u g0;
    g0[0] = 1u;                                   // count=1, user mode
    g0[1] = ldsoff;                               // lds_addr
    g0[2] = (unsigned)ga;                         // global_addr[31:0]
    g0[3] = (unsigned)((ga >> 32) & 0x1FFFFFFu)   // global_addr[56:32]
          | (2u << 30);                           // type = 2 ("image")
    v8i g1;
    g1[0] = (1 << 16)                             // data_size = 2 bytes
          | (1 << 20)                             // pad_enable
          | (4 << 22)                             // pad_interval: 32 dwords
          | (3 << 25);                            // pad_amount: 4 dwords
    g1[1] = (int)((dim0 & 0xFFFFu) << 16);        // tensor_dim0 lo16 in [63:48]
    g1[2] = (int)(((dim0 >> 16) & 0xFFFFu)        // tensor_dim0 hi16 in [79:64]
          | ((dim1 & 0xFFFFu) << 16));            // tensor_dim1 lo16 in [95:80]
    g1[3] = (int)(((dim1 >> 16) & 0xFFFFu)        // tensor_dim1 hi16 in [111:96]
          | (64u << 16));                         // tile_dim0 = 64 in [127:112]
    g1[4] = 64;                                   // tile_dim1 = 64 in [143:128]
    g1[5] = (int)rowStride;                       // tensor_dim0_stride lo32
    g1[6] = 0;                                    // stride hi16 + dim1_stride lo
    g1[7] = 0;
    v4i z4 = {0, 0, 0, 0};
#if __clang_major__ >= 23
    v8i z8 = {0, 0, 0, 0, 0, 0, 0, 0};
    __builtin_amdgcn_tensor_load_to_lds(g0, g1, z4, z4, z8, 0);
#else
    __builtin_amdgcn_tensor_load_to_lds(g0, g1, z4, z4, 0);
#endif
}
#endif

// ---------------------------------------------------------------------------
// Kernel 1: Y = X @ W^T  (f32 in, f16 out).
// VT=false: Y stored as [B, H, S, DH]  (Q and K)
// VT=true : Y stored as [B, H, DH, S]  (V, pre-transposed for the P*V GEMM)
// 128x64 output tile per workgroup, K-step 32.
// 4 waves x (32 rows x 64 cols) -> 2 A-frags, 8 accumulators, 8 WMMA/K-step.
// ---------------------------------------------------------------------------
template <bool VT>
__global__ __launch_bounds__(128) void qkv_proj_kernel(
    const float* __restrict__ X,     // [Msz, Ddim] row-major
    const float* __restrict__ W,     // [Ddim, Ddim] row-major; y = x @ W^T
    _Float16* __restrict__ Y)
{
    constexpr int TM = 128, TN = 64, BK = 32;
    __shared__ __align__(16) _Float16 As[TM][BK + 8];  // stride 40 halves = 80B
    __shared__ __align__(16) _Float16 Bs[TN][BK + 8];

    const int m0   = blockIdx.x * TM;
    const int n0   = blockIdx.y * TN;
    const int tid  = threadIdx.x;
    const int wave = tid >> 5;
    const int lane = tid & 31;
    const int l15  = lane & 15;
    const bool hi  = (lane >= 16);

    v8f acc[2][4] = {};

    for (int k0 = 0; k0 < Ddim; k0 += BK) {
        // Cooperative f32->f16 tile loads (float4 vectorized).
        for (int e = tid; e < TM * BK / 4; e += 128) {
            int r  = e >> 3;
            int c4 = (e & 7) * 4;
            float4 xa = *(const float4*)(X + (size_t)(m0 + r) * Ddim + k0 + c4);
            H4 ha; ha.h[0] = (_Float16)xa.x; ha.h[1] = (_Float16)xa.y;
                   ha.h[2] = (_Float16)xa.z; ha.h[3] = (_Float16)xa.w;
            *(uint2*)&As[r][c4] = ha.u;
        }
        for (int e = tid; e < TN * BK / 4; e += 128) {
            int r  = e >> 3;
            int c4 = (e & 7) * 4;
            float4 wb = *(const float4*)(W + (size_t)(n0 + r) * Ddim + k0 + c4);
            H4 hb; hb.h[0] = (_Float16)wb.x; hb.h[1] = (_Float16)wb.y;
                   hb.h[2] = (_Float16)wb.z; hb.h[3] = (_Float16)wb.w;
            *(uint2*)&Bs[r][c4] = hb.u;
        }
        // Prefetch next K-step's rows while WMMAs run on this one.
        if (k0 + BK < Ddim) {
            __builtin_prefetch(X + (size_t)(m0 + tid) * Ddim + k0 + BK, 0, 1);
            if (tid < TN)
                __builtin_prefetch(W + (size_t)(n0 + tid) * Ddim + k0 + BK, 0, 1);
        }
        __syncthreads();

        U16 a[2];
        #pragma unroll
        for (int mi = 0; mi < 2; ++mi) {
            int arow = wave * 32 + mi * 16 + l15;
            int kh   = hi ? 8 : 0;
            a[mi].q[0] = *(const uint4*)&As[arow][kh];
            a[mi].q[1] = *(const uint4*)&As[arow][kh + 16];
        }
        #pragma unroll
        for (int t = 0; t < 4; ++t) {
            U16 bfr;
            int bcol = t * 16 + l15;
            int kb   = hi ? 8 : 0;
            bfr.q[0] = *(const uint4*)&Bs[bcol][kb];
            bfr.q[1] = *(const uint4*)&Bs[bcol][kb + 16];
            #pragma unroll
            for (int mi = 0; mi < 2; ++mi)
                acc[mi][t] = __builtin_amdgcn_wmma_f32_16x16x32_f16(
                    false, a[mi].v, false, bfr.v, (short)0, acc[mi][t], false, false);
        }
        __syncthreads();
    }

    // Epilogue: scatter f16 into [B,H,S,DH] (Q/K) or [B,H,DH,S] (V).
    #pragma unroll
    for (int mi = 0; mi < 2; ++mi) {
        #pragma unroll
        for (int t = 0; t < 4; ++t) {
            int n  = n0 + t * 16 + l15;
            int h  = n >> 6;
            int dh = n & 63;
            #pragma unroll
            for (int r = 0; r < 8; ++r) {
                int m  = m0 + wave * 32 + mi * 16 + r + (hi ? 8 : 0);
                int bb = m >> 11;          // m / S
                int s  = m & (Sdim - 1);
                size_t off = VT
                    ? (((size_t)bb * Hn + h) * DHn + dh) * Sdim + s
                    : (((size_t)bb * Hn + h) * Sdim + s) * DHn + dh;
                Y[off] = (_Float16)acc[mi][t][r];
            }
        }
    }
}

// ---------------------------------------------------------------------------
// Kernel 2: flash attention. Q,K in [B,H,S,DH] f16; V pre-transposed in
// [B,H,DH,S] f16. Workgroup = (q-tile of 64 rows) x (one b,h); 4 waves x
// 16 query rows. Both K and V^T tiles are DMA'd by the Tensor Data Mover
// (wave 0 and wave 1 issue in parallel) into padded LDS. Fragment bursts are
// preloaded into distinct registers so the 8-WMMA sequences issue without
// per-WMMA dscnt stalls.
// ---------------------------------------------------------------------------
__global__ __launch_bounds__(128) void flash_attn_kernel(
    const _Float16* __restrict__ Qm,      // [B,H,S,DH]
    const _Float16* __restrict__ Km,      // [B,H,S,DH]
    const _Float16* __restrict__ Vt,      // [B,H,DH,S]
    const int* __restrict__ pad,          // [B, S]
    float* __restrict__ Out)              // [B, S, D]
{
    constexpr int BC = 64;                               // keys per chunk
    __shared__ __align__(16) _Float16 Ks [BC][DHn + 8];  // [key][dh], 144B rows
    __shared__ __align__(16) _Float16 Vst[DHn][BC + 8];  // [dh][key], 144B rows
    __shared__ __align__(16) _Float16 Ps [4][16][BC + 8];// per-wave P staging

    const int qt  = blockIdx.x;            // 0..31
    const int bh  = blockIdx.y;            // 0..31
    const int b   = bh >> 4;
    const int h   = bh & 15;
    const int tid = threadIdx.x;
    const int wave = tid >> 5;
    const int lane = tid & 31;
    const int l15  = lane & 15;
    const bool hiL = (lane >= 16);
    const int q0   = qt * 64;
    const int myq  = q0 + wave * 16;
    const size_t headOff  = ((size_t)b * Hn + h) * Sdim;        // rows of Q/K
    const size_t vHeadOff = ((size_t)b * Hn + h) * DHn * Sdim;  // V^T block

    // Load this wave's Q fragments once (2 chunks of dh), folding in 1/sqrt(DH).
    U16 qa[2];
    {
        const _Float16* qp = Qm + (headOff + (size_t)(myq + l15)) * DHn;
        int khalf = hiL ? 8 : 0;
        #pragma unroll
        for (int ks = 0; ks < 2; ++ks) {
            qa[ks].q[0] = *(const uint4*)(qp + ks * 32 + khalf);
            qa[ks].q[1] = *(const uint4*)(qp + ks * 32 + khalf + 16);
            #pragma unroll
            for (int e = 0; e < 16; ++e)
                qa[ks].h[e] = qa[ks].h[e] * (_Float16)0.125f;   // 1/sqrt(64)
        }
    }

    v8f o[4] = {};
    float m_prev[8], l_prev[8];
    #pragma unroll
    for (int r = 0; r < 8; ++r) { m_prev[r] = -3.0e38f; l_prev[r] = 0.0f; }

    for (int kc = 0; kc <= qt; ++kc) {
        const int kbase = kc * 64;

        // ---- K and V^T tiles into padded LDS ------------------------------
#if HAS_TDM
        if (wave == 0) {
            tdm_load_tile64(Km + (headOff + kbase) * DHn,
                            (unsigned)(size_t)(void*)&Ks[0][0],
                            DHn, DHn, (unsigned)(Sdim - kbase));
        } else if (wave == 1) {
            tdm_load_tile64(Vt + vHeadOff + kbase,
                            (unsigned)(size_t)(void*)&Vst[0][0],
                            Sdim, (unsigned)(Sdim - kbase), DHn);
        }
#elif HAS_ASYNC
        for (int e = tid; e < BC * DHn / 8; e += 128) {
            int r = e >> 3, c = (e & 7) * 8;
            __builtin_amdgcn_global_load_async_to_lds_b128(
                (__attribute__((address_space(1))) void*)(Km + (headOff + kbase + r) * DHn + c),
                (__attribute__((address_space(3))) void*)&Ks[r][c], 0, 0);
            __builtin_amdgcn_global_load_async_to_lds_b128(
                (__attribute__((address_space(1))) void*)(Vt + vHeadOff + (size_t)r * Sdim + kbase + c),
                (__attribute__((address_space(3))) void*)&Vst[r][c], 0, 0);
        }
#else
        for (int e = tid; e < BC * DHn / 8; e += 128) {
            int r = e >> 3, c = (e & 7) * 8;
            *(uint4*)&Ks[r][c] =
                *(const uint4*)(Km + (headOff + kbase + r) * DHn + c);
            *(uint4*)&Vst[r][c] =
                *(const uint4*)(Vt + vHeadOff + (size_t)r * Sdim + kbase + c);
        }
#endif

        // Prefetch next key chunk while this one computes.
        if (kc < qt) {
            const int nb = kbase + BC;
            if (tid < BC)
                __builtin_prefetch(Km + (headOff + nb + tid) * DHn, 0, 1);
            else
                __builtin_prefetch(Vt + vHeadOff + (size_t)(tid - BC) * Sdim + nb, 0, 1);
        }

#if HAS_TDM
        if (wave < 2) __builtin_amdgcn_s_wait_tensorcnt(0);
#elif HAS_ASYNC
#if HAS_WAIT_ASYNC
        __builtin_amdgcn_s_wait_asynccnt(0);
#else
        asm volatile("s_wait_asynccnt 0" ::: "memory");
#endif
#endif
        __syncthreads();

        // ---- Issue pad-mask loads early (latency hides under WMMA burst) --
        int pm[4];
        #pragma unroll
        for (int t = 0; t < 4; ++t)
            pm[t] = pad[b * Sdim + kbase + t * 16 + l15];

        // ---- Scores: S = Q * K^T (preload all 8 B-frags, then 8 WMMAs) ----
        U16 bf[8];
        #pragma unroll
        for (int i = 0; i < 8; ++i) {
            int t = i >> 1, ks = i & 1;
            const _Float16* bp = &Ks[t * 16 + l15][ks * 32 + (hiL ? 8 : 0)];
            bf[i].q[0] = *(const uint4*)bp;
            bf[i].q[1] = *(const uint4*)(bp + 16);
        }
        v8f sc[4];
        #pragma unroll
        for (int t = 0; t < 4; ++t) {
            v8f z = {};
            z = __builtin_amdgcn_wmma_f32_16x16x32_f16(
                false, qa[0].v, false, bf[2 * t].v, (short)0, z, false, false);
            sc[t] = __builtin_amdgcn_wmma_f32_16x16x32_f16(
                false, qa[1].v, false, bf[2 * t + 1].v, (short)0, z, false, false);
        }

        // ---- Masking: pad mask always; causal only on diagonal chunk ------
        const bool diag = (kc == qt);
        #pragma unroll
        for (int t = 0; t < 4; ++t) {
            int key = kbase + t * 16 + l15;
            #pragma unroll
            for (int r = 0; r < 8; ++r) {
                int qrow = myq + r + (hiL ? 8 : 0);
                float s = sc[t][r];
                if (pm[t] == 0 || (diag && key > qrow)) s = NEGV;
                sc[t][r] = s;
            }
        }

        // ---- Online softmax (rows span 16 lanes of a half; wave32) --------
        #pragma unroll
        for (int r = 0; r < 8; ++r) {
            float mx = fmaxf(fmaxf(sc[0][r], sc[1][r]), fmaxf(sc[2][r], sc[3][r]));
            #pragma unroll
            for (int off = 1; off < 16; off <<= 1)
                mx = fmaxf(mx, __shfl_xor(mx, off, 32));
            float mi = fmaxf(m_prev[r], mx);
            float rowsum = 0.0f;
            #pragma unroll
            for (int t = 0; t < 4; ++t) {
                float p = __expf(sc[t][r] - mi);
                sc[t][r] = p;
                rowsum  += p;
            }
            #pragma unroll
            for (int off = 1; off < 16; off <<= 1)
                rowsum += __shfl_xor(rowsum, off, 32);
            float alpha = __expf(m_prev[r] - mi);
            l_prev[r] = l_prev[r] * alpha + rowsum;
            m_prev[r] = mi;
            #pragma unroll
            for (int t = 0; t < 4; ++t) o[t][r] *= alpha;
        }

        // ---- Re-layout P (C-layout -> A-layout) through LDS staging -------
        #pragma unroll
        for (int t = 0; t < 4; ++t) {
            int col = t * 16 + l15;
            #pragma unroll
            for (int r = 0; r < 8; ++r)
                Ps[wave][r + (hiL ? 8 : 0)][col] = (_Float16)sc[t][r];
        }

        // ---- O += P * V: preload pa[2] + all 8 V-frags, then 8 WMMAs ------
        U16 pa[2];
        #pragma unroll
        for (int ks = 0; ks < 2; ++ks) {
            const _Float16* pp = &Ps[wave][l15][ks * 32 + (hiL ? 8 : 0)];
            pa[ks].q[0] = *(const uint4*)pp;
            pa[ks].q[1] = *(const uint4*)(pp + 16);
        }
        U16 vf[8];
        #pragma unroll
        for (int i = 0; i < 8; ++i) {
            int t = i >> 1, ks = i & 1;
            const _Float16* vp = &Vst[t * 16 + l15][ks * 32 + (hiL ? 8 : 0)];
            vf[i].q[0] = *(const uint4*)vp;
            vf[i].q[1] = *(const uint4*)(vp + 16);
        }
        #pragma unroll
        for (int t = 0; t < 4; ++t) {
            v8f tmp;
            tmp  = __builtin_amdgcn_wmma_f32_16x16x32_f16(
                false, pa[0].v, false, vf[2 * t].v, (short)0, o[t], false, false);
            o[t] = __builtin_amdgcn_wmma_f32_16x16x32_f16(
                false, pa[1].v, false, vf[2 * t + 1].v, (short)0, tmp, false, false);
        }
        __syncthreads();   // before next chunk overwrites Ks/Vst
    }

    // ---- Epilogue: out[b, q, h*64 + dh] = o / l ---------------------------
    #pragma unroll
    for (int t = 0; t < 4; ++t) {
        int dh = t * 16 + l15;
        #pragma unroll
        for (int r = 0; r < 8; ++r) {
            int row = myq + r + (hiL ? 8 : 0);
            Out[((size_t)b * Sdim + row) * Ddim + h * DHn + dh] =
                o[t][r] / l_prev[r];
        }
    }
}

// ---------------------------------------------------------------------------
// Launch: 3 projection GEMMs into d_ws (f16, 24 MB total; V pre-transposed),
// then one flash-attention kernel producing the f32 output.
// ---------------------------------------------------------------------------
extern "C" void kernel_launch(void* const* d_in, const int* in_sizes, int n_in,
                              void* d_out, int out_size, void* d_ws, size_t ws_size,
                              hipStream_t stream) {
    const float* query = (const float*)d_in[0];
    const float* key   = (const float*)d_in[1];
    const float* value = (const float*)d_in[2];
    const float* Wq    = (const float*)d_in[3];
    const float* Wk    = (const float*)d_in[4];
    const float* Wv    = (const float*)d_in[5];
    const int*   pad   = (const int*)d_in[6];
    // d_in[7] = attn_mask (causal tril) is handled analytically in-kernel.

    const size_t perTensor = (size_t)Bn * Hn * Sdim * DHn;  // 4 Mi f16 elems
    _Float16* Qw = (_Float16*)d_ws;
    _Float16* Kw = Qw + perTensor;
    _Float16* Vw = Kw + perTensor;      // stored [B,H,DH,S]

    dim3 g1(Msz / 128, Ddim / 64), b1(128);
    qkv_proj_kernel<false><<<g1, b1, 0, stream>>>(query, Wq, Qw);
    qkv_proj_kernel<false><<<g1, b1, 0, stream>>>(key,   Wk, Kw);
    qkv_proj_kernel<true ><<<g1, b1, 0, stream>>>(value, Wv, Vw);

    dim3 g2(Sdim / 64, Bn * Hn), b2(128);
    flash_attn_kernel<<<g2, b2, 0, stream>>>(Qw, Kw, Vw, pad, (float*)d_out);
}
